// SE3Backbone_16913581211670
// MI455X (gfx1250) — matrix operand order repaired and checked by reading, not compiled, not run on hardware
//
#include <hip/hip_runtime.h>
#include <math.h>

#define NN 20000
#define EE 320000
#define FD 128
#define RB 32
#define NBK 3
#define CUT 5.0f
#define MIND 0.3f
#define PIF 3.14159265358979f

typedef __attribute__((ext_vector_type(16))) __bf16 bf16x16;
typedef __attribute__((ext_vector_type(8)))  float  f32x8;

// fast swish: x * rcp(1+exp(-x)) -> v_exp_f32 + v_rcp_f32 instead of precise-div sequence
__device__ __forceinline__ float swishf(float x) {
  return x * __builtin_amdgcn_rcpf(1.0f + __expf(-x));
}

__device__ __forceinline__ void fatomic_add(float* p, float v) {
  __hip_atomic_fetch_add(p, v, __ATOMIC_RELAXED, __HIP_MEMORY_SCOPE_AGENT);
}

// A fragment (16x32 bf16 window) from row-major shared/global memory with leading dim ld.
// Layout (ISA 7.12.2): lanes 0-15: M=lane, K={0..7,16..23}; lanes 16-31: M=lane-16, K={8..15,24..31}
__device__ __forceinline__ bf16x16 frag_a(const __bf16* a, int ld) {
  int lane = threadIdx.x & 31;
  const __bf16* p = a + (lane & 15) * ld + ((lane >> 4) << 3);
  bf16x16 f;
#pragma unroll
  for (int e = 0; e < 8; ++e) { f[e] = p[e]; f[e + 8] = p[e + 16]; }
  return f;
}

// B fragment (32x16 bf16 window) from column-major weights, leading dim Kld.
// lane L holds column n0+(L&15), K = k0 + 8*(L>=16) + {0..7,16..23}
__device__ __forceinline__ bf16x16 frag_b(const __bf16* w, int Kld, int n0, int k0) {
  int lane = threadIdx.x & 31;
  const __bf16* p = w + (size_t)(n0 + (lane & 15)) * Kld + k0 + ((lane >> 4) << 3);
  bf16x16 f;
#pragma unroll
  for (int e = 0; e < 8; ++e) { f[e] = p[e]; f[e + 8] = p[e + 16]; }
  return f;
}

#define WMMA(a, b, c) __builtin_amdgcn_wmma_f32_16x16x32_bf16(false, (a), false, (b), (short)0, (c), false, false)

// ---------------- utility kernels ----------------

__global__ void k_cvt(const float* __restrict__ in, __bf16* __restrict__ out, int K, int Nw) {
  int idx = blockIdx.x * 256 + threadIdx.x;
  if (idx >= K * Nw) return;
  int k = idx / Nw, n = idx - k * Nw;
  out[(size_t)n * K + k] = (__bf16)in[idx];   // row-major (K x N) -> column-major
}

__global__ void k_zero(float* __restrict__ p, long n) {
  long i = (long)blockIdx.x * 256 + threadIdx.x;
  if (i < n) p[i] = 0.0f;
}

// ---------------- per-edge geometry ----------------

__global__ __launch_bounds__(256) void k_geom(
    const float* __restrict__ pos, const int* __restrict__ ei, const int* __restrict__ ej,
    const float* __restrict__ freq,
    float* __restrict__ sh8, float* __restrict__ maskA, __bf16* __restrict__ rbf)
{
  int e = blockIdx.x * 256 + threadIdx.x;
  if (e >= EE) return;
  int i = ei[e], j = ej[e];
  float dx = pos[j * 3 + 0] - pos[i * 3 + 0];
  float dy = pos[j * 3 + 1] - pos[i * 3 + 1];
  float dz = pos[j * 3 + 2] - pos[i * 3 + 2];
  float r = sqrtf(dx * dx + dy * dy + dz * dz + 1e-12f);
  float inv = 1.0f / (r + 1e-6f);
  float x = dx * inv, y = dy * inv, z = dz * inv;
  float mask = (r >= MIND && r < CUT) ? 1.0f : 0.0f;
  float fc = 0.5f * (__cosf(PIF * r / CUT) + 1.0f) * mask;
  const float s3 = 1.7320508075688772f, s5 = 2.23606797749979f, s15 = 3.872983346207417f;
  float* sp = sh8 + (size_t)e * 8;
  sp[0] = s3 * y; sp[1] = s3 * z; sp[2] = s3 * x;
  sp[3] = s15 * x * y; sp[4] = s15 * y * z; sp[5] = 0.5f * s5 * (3.0f * z * z - 1.0f);
  sp[6] = s15 * z * x; sp[7] = 0.5f * s15 * (x * x - y * y);
  maskA[e] = mask;
  float c0 = sqrtf(2.0f / CUT) * inv * fc;
  __bf16* rp = rbf + (size_t)e * RB;
#pragma unroll
  for (int k = 0; k < RB; ++k)
    rp[k] = (__bf16)(c0 * __sinf(freq[k] * r / CUT));
}

// ---------------- node embedding: S = swish(emb[z] @ Wemb + b) ----------------

__global__ __launch_bounds__(256) void k_embed(
    const int* __restrict__ z, const float* __restrict__ emb, const __bf16* __restrict__ Wt,
    const float* __restrict__ bias, float* __restrict__ S, __bf16* __restrict__ Sbf)
{
  __shared__ __attribute__((aligned(32))) __bf16 A[16 * FD];
  int t = threadIdx.x, wv = t >> 5, lane = t & 31;
  int n0row = blockIdx.x * 16;
  { int r = t >> 4, c0 = (t & 15) * 8;
    const float* src = emb + (size_t)z[n0row + r] * FD + c0;
#pragma unroll
    for (int q = 0; q < 8; ++q) A[r * FD + c0 + q] = (__bf16)src[q];
  }
  __syncthreads();
  int n0 = wv * 16;
  f32x8 acc = {};
#pragma unroll
  for (int k0 = 0; k0 < FD; k0 += 32) acc = WMMA(frag_a(A + k0, FD), frag_b(Wt, FD, n0, k0), acc);
  int col = n0 + (lane & 15);
  float b = bias[col];
#pragma unroll
  for (int k = 0; k < 8; ++k) {
    int m = k + ((lane >> 4) << 3);
    float v = swishf(acc[k] + b);
    size_t idx = (size_t)(n0row + m) * FD + col;
    S[idx] = v; Sbf[idx] = (__bf16)v;
  }
}

// ---------------- fused edge kernel: filter MLP + Wsrc + Wu + messages + scatter ----------------

__global__ __launch_bounds__(256) void k_edge(
    const int* __restrict__ ei, const int* __restrict__ ej,
    const __bf16* __restrict__ rbf, const float* __restrict__ sh8,
    const float* __restrict__ maskA, const __bf16* __restrict__ Sbf,
    const __bf16* __restrict__ Wf1t, const float* __restrict__ b1,
    const __bf16* __restrict__ Wf2t, const __bf16* __restrict__ Wsrct,
    const __bf16* __restrict__ Wut,
    float* __restrict__ aggS, float* __restrict__ aggT)
{
  __shared__ __attribute__((aligned(32))) __bf16 Argb[16 * RB];
  __shared__ __attribute__((aligned(32))) __bf16 Ahid[16 * FD];
  __shared__ __attribute__((aligned(32))) __bf16 Asj[16 * FD];
  __shared__ int sI[16];
  __shared__ float sSh[16 * 8];
  __shared__ float sMask[16];
  int t = threadIdx.x, wv = t >> 5, lane = t & 31;
  size_t e0 = (size_t)blockIdx.x * 16;
  if (t < 16) { sI[t] = ei[e0 + t]; sMask[t] = maskA[e0 + t]; }
  if (t < 128) sSh[t] = sh8[e0 * 8 + t];
  { int r = t >> 4, c0 = (t & 15) * 2;
    Argb[r * RB + c0]     = rbf[(e0 + r) * RB + c0];
    Argb[r * RB + c0 + 1] = rbf[(e0 + r) * RB + c0 + 1]; }
  { int r = t >> 4, c0 = (t & 15) * 8;
    int j = ej[e0 + r];
    const __bf16* src = Sbf + (size_t)j * FD + c0;
#pragma unroll
    for (int q = 0; q < 8; ++q) Asj[r * FD + c0 + q] = src[q]; }
  __syncthreads();

  int n0 = wv * 16;
  int col = n0 + (lane & 15);

  // GEMM1: hidden = swish(rbf @ Wf1 + b1)   (K=32)
  f32x8 h = {};
  h = WMMA(frag_a(Argb, RB), frag_b(Wf1t, RB, n0, 0), h);
  float bb = b1[col];
#pragma unroll
  for (int k = 0; k < 8; ++k) {
    int m = k + ((lane >> 4) << 3);
    Ahid[m * FD + col] = (__bf16)swishf(h[k] + bb);
  }
  __syncthreads();

  // GEMM2: w = hidden @ Wf2 (512 wide) -> wS,w1,w2,w3 tiles
  f32x8 wacc[4] = { {}, {}, {}, {} };
#pragma unroll
  for (int tgt = 0; tgt < 4; ++tgt)
#pragma unroll
    for (int k0 = 0; k0 < FD; k0 += 32)
      wacc[tgt] = WMMA(frag_a(Ahid + k0, FD), frag_b(Wf2t, FD, tgt * FD + n0, k0), wacc[tgt]);

  // GEMM3: src = Sj @ Wsrc
  f32x8 cacc = {};
#pragma unroll
  for (int k0 = 0; k0 < FD; k0 += 32)
    cacc = WMMA(frag_a(Asj + k0, FD), frag_b(Wsrct, FD, n0, k0), cacc);

  // GEMM4: u = Sj @ Wu (384 wide) -> u1,u2,u3 tiles
  f32x8 uacc[3] = { {}, {}, {} };
#pragma unroll
  for (int tgt = 0; tgt < 3; ++tgt)
#pragma unroll
    for (int k0 = 0; k0 < FD; k0 += 32)
      uacc[tgt] = WMMA(frag_a(Asj + k0, FD), frag_b(Wut, FD, tgt * FD + n0, k0), uacc[tgt]);

  // epilogue: build messages and scatter-add (fp32 HW atomics, L2-resident)
#pragma unroll
  for (int k = 0; k < 8; ++k) {
    int m = k + ((lane >> 4) << 3);
    float msk = sMask[m];
    int node = sI[m];
    float sv = swishf(cacc[k]);
    fatomic_add(aggS + (size_t)node * FD + col, wacc[0][k] * msk * sv);
    float m1 = wacc[1][k] * msk * uacc[0][k];
    float m2 = wacc[2][k] * msk * uacc[1][k];
    float m3 = wacc[3][k] * msk * uacc[2][k];
    float* Tp = aggT + ((size_t)node * 11) * FD + col;
    const float* sp = sSh + m * 8;
#pragma unroll
    for (int d = 0; d < 3; ++d) fatomic_add(Tp + d * FD, sp[d] * m1);
#pragma unroll
    for (int d = 0; d < 3; ++d) fatomic_add(Tp + (3 + d) * FD, sp[d] * m2);
#pragma unroll
    for (int d = 0; d < 5; ++d) fatomic_add(Tp + (6 + d) * FD, sp[3 + d] * m3);
  }
}

// ---------------- node einsums: T[:,grp] = (T+aggT)[:,grp] @ W ----------------

__global__ __launch_bounds__(256) void k_nodeT(
    const float* __restrict__ aggT, float* __restrict__ T,
    const __bf16* __restrict__ Wt, int cg, int choff)
{
  __shared__ __attribute__((aligned(32))) __bf16 A[16 * FD];
  int t = threadIdx.x, wv = t >> 5, lane = t & 31;
  long row0 = (long)blockIdx.x * 16;
  { int r = t >> 4, c0 = (t & 15) * 8;
    long row = row0 + r;
    int node = (int)(row / cg);
    int ch = choff + (int)(row % cg);
    size_t base = ((size_t)node * 11 + ch) * FD + c0;
#pragma unroll
    for (int q = 0; q < 8; ++q) A[r * FD + c0 + q] = (__bf16)(T[base + q] + aggT[base + q]);
  }
  __syncthreads();
  int n0 = wv * 16;
  f32x8 acc = {};
#pragma unroll
  for (int k0 = 0; k0 < FD; k0 += 32) acc = WMMA(frag_a(A + k0, FD), frag_b(Wt, FD, n0, k0), acc);
  int col = n0 + (lane & 15);
#pragma unroll
  for (int k = 0; k < 8; ++k) {
    int m = k + ((lane >> 4) << 3);
    long row = row0 + m;
    int node = (int)(row / cg);
    int ch = choff + (int)(row % cg);
    T[((size_t)node * 11 + ch) * FD + col] = acc[k];
  }
}

// ---------------- invariants + U = [aggS | inv] (bf16) ----------------

__global__ void k_buildU(const float* __restrict__ T, const float* __restrict__ aggS,
                         __bf16* __restrict__ U)
{
  int idx = blockIdx.x * 256 + threadIdx.x;  // exactly NN*FD threads
  int node = idx / FD, col = idx - node * FD;
  const float* Tp = T + (size_t)node * 11 * FD + col;
  float i1 = 0.f, i2 = 0.f, i3 = 0.f;
#pragma unroll
  for (int d = 0; d < 3; ++d)  { float v = Tp[d * FD]; i1 += v * v; }
#pragma unroll
  for (int d = 3; d < 6; ++d)  { float v = Tp[d * FD]; i2 += v * v; }
#pragma unroll
  for (int d = 6; d < 11; ++d) { float v = Tp[d * FD]; i3 += v * v; }
  __bf16* u = U + (size_t)node * 4 * FD;
  u[col]          = (__bf16)aggS[(size_t)node * FD + col];
  u[FD + col]     = (__bf16)i1;
  u[2 * FD + col] = (__bf16)i2;
  u[3 * FD + col] = (__bf16)i3;
}

// ---------------- S += swish(U @ Wup)  (K=512) ----------------

__global__ __launch_bounds__(256) void k_update(
    const __bf16* __restrict__ U, const __bf16* __restrict__ Wupt,
    float* __restrict__ S, __bf16* __restrict__ Sbf)
{
  __shared__ __attribute__((aligned(32))) __bf16 A[16 * 4 * FD];
  int t = threadIdx.x, wv = t >> 5, lane = t & 31;
  int n0row = blockIdx.x * 16;
  { int r = t >> 4, c0 = (t & 15) * 32;
    const __bf16* src = U + (size_t)(n0row + r) * 4 * FD + c0;
#pragma unroll
    for (int q = 0; q < 32; ++q) A[r * 4 * FD + c0 + q] = src[q];
  }
  __syncthreads();
  int n0 = wv * 16;
  f32x8 acc = {};
#pragma unroll
  for (int k0 = 0; k0 < 4 * FD; k0 += 32)
    acc = WMMA(frag_a(A + k0, 4 * FD), frag_b(Wupt, 4 * FD, n0, k0), acc);
  int col = n0 + (lane & 15);
#pragma unroll
  for (int k = 0; k < 8; ++k) {
    int m = k + ((lane >> 4) << 3);
    size_t idx = (size_t)(n0row + m) * FD + col;
    float v = S[idx] + swishf(acc[k]);
    S[idx] = v; Sbf[idx] = (__bf16)v;
  }
}

// ---------------- final output copy ----------------

__global__ void k_out(const float* __restrict__ S, const float* __restrict__ T,
                      float* __restrict__ out)
{
  long idx = (long)blockIdx.x * 256 + threadIdx.x;
  const long ns = (long)NN * FD;
  const long total = ns + (long)NN * 11 * FD;
  if (idx >= total) return;
  out[idx] = (idx < ns) ? S[idx] : T[idx - ns];
}

// ---------------- host launcher ----------------

extern "C" void kernel_launch(void* const* d_in, const int* in_sizes, int n_in,
                              void* d_out, int out_size, void* d_ws, size_t ws_size,
                              hipStream_t stream) {
  const int*   z      = (const int*)d_in[0];
  const float* pos    = (const float*)d_in[1];
  const int*   eidx   = (const int*)d_in[2];
  const float* emb    = (const float*)d_in[3];
  const float* emb_W  = (const float*)d_in[4];
  const float* emb_b  = (const float*)d_in[5];
  const float* freq   = (const float*)d_in[6];
  const float* Wf1    = (const float*)d_in[7];
  const float* bf1    = (const float*)d_in[8];
  const float* Wf2    = (const float*)d_in[9];
  const float* Wsrc   = (const float*)d_in[10];
  const float* Wu     = (const float*)d_in[11];
  const float* W1o    = (const float*)d_in[12];
  const float* W1e    = (const float*)d_in[13];
  const float* W2e    = (const float*)d_in[14];
  const float* Wup    = (const float*)d_in[15];

  char* p = (char*)d_ws;
  auto carve = [&](size_t bytes) -> void* {
    void* r = (void*)p; p += (bytes + 255) & ~(size_t)255; return r;
  };
  __bf16* WembT = (__bf16*)carve((size_t)FD * FD * 2);
  __bf16* Wf1T  = (__bf16*)carve((size_t)NBK * RB * FD * 2);
  __bf16* Wf2T  = (__bf16*)carve((size_t)NBK * FD * 4 * FD * 2);
  __bf16* WsrcT = (__bf16*)carve((size_t)NBK * FD * FD * 2);
  __bf16* WuT   = (__bf16*)carve((size_t)NBK * FD * 3 * FD * 2);
  __bf16* W1oT  = (__bf16*)carve((size_t)NBK * FD * FD * 2);
  __bf16* W1eT  = (__bf16*)carve((size_t)NBK * FD * FD * 2);
  __bf16* W2eT  = (__bf16*)carve((size_t)NBK * FD * FD * 2);
  __bf16* WupT  = (__bf16*)carve((size_t)NBK * 4 * FD * FD * 2);
  float*  sh8   = (float*)carve((size_t)EE * 8 * 4);
  float*  maskA = (float*)carve((size_t)EE * 4);
  __bf16* rbfB  = (__bf16*)carve((size_t)EE * RB * 2);
  float*  S     = (float*)carve((size_t)NN * FD * 4);
  __bf16* Sbf   = (__bf16*)carve((size_t)NN * FD * 2);
  float*  T     = (float*)carve((size_t)NN * 11 * FD * 4);
  float*  aggS  = (float*)carve((size_t)NN * FD * 4);
  float*  aggT  = (float*)carve((size_t)NN * 11 * FD * 4);
  __bf16* U     = (__bf16*)carve((size_t)NN * 4 * FD * 2);

  auto cvt = [&](const float* in, __bf16* out, int K, int Nw) {
    int n = K * Nw;
    k_cvt<<<(n + 255) / 256, 256, 0, stream>>>(in, out, K, Nw);
  };
  cvt(emb_W, WembT, FD, FD);
  for (int b = 0; b < NBK; ++b) {
    cvt(Wf1 + (size_t)b * RB * FD,       Wf1T + (size_t)b * RB * FD,       RB, FD);
    cvt(Wf2 + (size_t)b * FD * 4 * FD,   Wf2T + (size_t)b * FD * 4 * FD,   FD, 4 * FD);
    cvt(Wsrc + (size_t)b * FD * FD,      WsrcT + (size_t)b * FD * FD,      FD, FD);
    cvt(Wu + (size_t)b * FD * 3 * FD,    WuT + (size_t)b * FD * 3 * FD,    FD, 3 * FD);
    cvt(W1o + (size_t)b * FD * FD,       W1oT + (size_t)b * FD * FD,       FD, FD);
    cvt(W1e + (size_t)b * FD * FD,       W1eT + (size_t)b * FD * FD,       FD, FD);
    cvt(W2e + (size_t)b * FD * FD,       W2eT + (size_t)b * FD * FD,       FD, FD);
    cvt(Wup + (size_t)b * 4 * FD * FD,   WupT + (size_t)b * 4 * FD * FD,   4 * FD, FD);
  }

  k_geom<<<(EE + 255) / 256, 256, 0, stream>>>(pos, eidx, eidx + EE, freq, sh8, maskA, rbfB);

  long nT = (long)NN * 11 * FD;
  k_zero<<<(int)((nT + 255) / 256), 256, 0, stream>>>(T, nT);

  k_embed<<<NN / 16, 256, 0, stream>>>(z, emb, WembT, emb_b, S, Sbf);

  for (int b = 0; b < NBK; ++b) {
    long nS = (long)NN * FD;
    k_zero<<<(int)((nS + 255) / 256), 256, 0, stream>>>(aggS, nS);
    k_zero<<<(int)((nT + 255) / 256), 256, 0, stream>>>(aggT, nT);
    k_edge<<<EE / 16, 256, 0, stream>>>(
        eidx, eidx + EE, rbfB, sh8, maskA, Sbf,
        Wf1T + (size_t)b * RB * FD, bf1 + (size_t)b * FD,
        Wf2T + (size_t)b * FD * 4 * FD, WsrcT + (size_t)b * FD * FD,
        WuT + (size_t)b * FD * 3 * FD, aggS, aggT);
    k_nodeT<<<(NN * 3) / 16, 256, 0, stream>>>(aggT, T, W1oT + (size_t)b * FD * FD, 3, 0);
    k_nodeT<<<(NN * 3) / 16, 256, 0, stream>>>(aggT, T, W1eT + (size_t)b * FD * FD, 3, 3);
    k_nodeT<<<(NN * 5) / 16, 256, 0, stream>>>(aggT, T, W2eT + (size_t)b * FD * FD, 5, 6);
    k_buildU<<<(NN * FD) / 256, 256, 0, stream>>>(T, aggS, U);
    k_update<<<NN / 16, 256, 0, stream>>>(U, WupT + (size_t)b * 4 * FD * FD, S, Sbf);
  }

  long total = (long)NN * FD + (long)NN * 11 * FD;
  k_out<<<(int)((total + 255) / 256), 256, 0, stream>>>(S, T, (float*)d_out);
}
